// DotProductAttention_34351148433410
// MI455X (gfx1250) — matrix operand order repaired and checked by reading, not compiled, not run on hardware
//
#include <hip/hip_runtime.h>

// ---------------- problem constants (match reference) ----------------
constexpr int Hd = 1024;          // hidden size H
constexpr int K2 = 2048;          // 2H
constexpr int Bb = 8;             // batch
constexpr int Ss = 4096;          // encoder seq len S
constexpr int Ll = 64;            // decoder len L

typedef unsigned short u16;
typedef __attribute__((ext_vector_type(16))) __bf16        v16bf;
typedef __attribute__((ext_vector_type(8)))  float         v8f;
typedef __attribute__((ext_vector_type(8)))  unsigned int  v8u;
typedef int v4i __attribute__((vector_size(16)));          // builtin's b128 type

#define AS1 __attribute__((address_space(1)))
#define AS3 __attribute__((address_space(3)))

// ---- gfx1250 async global->LDS path (probe-confirmed to exist) ----
#if __has_builtin(__builtin_amdgcn_global_load_async_to_lds_b128)
#define HAVE_ASYNC_LDS 1
#endif

#if defined(HAVE_ASYNC_LDS)
#if __has_builtin(__builtin_amdgcn_s_wait_asynccnt)
#define ASYNC_WAIT() __builtin_amdgcn_s_wait_asynccnt(0)
#else
#define ASYNC_WAIT() asm volatile("s_wait_asynccnt 0" ::: "memory")
#endif
#else
#define ASYNC_WAIT()
#endif

// ---------------- helpers ----------------
__device__ __forceinline__ u16 f2bf(float f) {           // fp32 -> bf16 RNE
    unsigned int u = __float_as_uint(f);
    unsigned int r = u + 0x7FFFu + ((u >> 16) & 1u);
    return (u16)(r >> 16);
}
__device__ __forceinline__ unsigned int pk2(float lo, float hi) {
    return (unsigned int)f2bf(lo) | ((unsigned int)f2bf(hi) << 16);
}

// A-matrix fragment (16x32 bf16): per lane 16 contiguous K starting at koff
// (koff = (lane>>4)*16); caller passes p = row + koff.
__device__ __forceinline__ v16bf frag_a16(const u16* p) {
    uint4 a = *reinterpret_cast<const uint4*>(p);
    uint4 b = *reinterpret_cast<const uint4*>(p + 8);
    v8u u = {a.x, a.y, a.z, a.w, b.x, b.y, b.z, b.w};
    return __builtin_bit_cast(v16bf, u);
}
// B-matrix fragment (32x16 bf16): per lane 16 contiguous K starting at koff
// (koff = (lane>>4)*16); caller passes p = column-base + koff.
__device__ __forceinline__ v16bf frag_b16(const u16* p) {
    uint4 a = *reinterpret_cast<const uint4*>(p);
    uint4 b = *reinterpret_cast<const uint4*>(p + 8);
    v8u u = {a.x, a.y, a.z, a.w, b.x, b.y, b.z, b.w};
    return __builtin_bit_cast(v16bf, u);
}
__device__ __forceinline__ v8f wmma_bf16(v16bf a, v16bf b, v8f c) {
    return __builtin_amdgcn_wmma_f32_16x16x32_bf16(false, a, false, b, (short)0, c,
                                                   false, false);
}

// ---------------- kernel 1: fp32 -> bf16 converts (W, hidden) ----------------
__global__ __launch_bounds__(256) void convert_kernel(const float* __restrict__ Wf,
                                                      u16* __restrict__ Wb, int nW,
                                                      const float* __restrict__ Hf,
                                                      u16* __restrict__ Hb, int nH) {
    int stride = gridDim.x * blockDim.x;
    for (int i = blockIdx.x * blockDim.x + threadIdx.x; i < nW; i += stride)
        Wb[i] = f2bf(Wf[i]);
    for (int i = blockIdx.x * blockDim.x + threadIdx.x; i < nH; i += stride)
        Hb[i] = f2bf(Hf[i]);
}

// ---------------- kernel 2: reduced = enc @ W^T + b  (bf16 out) ----------------
// M = B*S = 32768, K = 2H = 2048, N = H = 1024. 128x128 C tile per block,
// 8 waves, each wave 64x32 subtile (4x2 WMMA frags), K-step 32.
// Double-buffered LDS; B tile staged via async global->LDS (ASYNCcnt).
__global__ __launch_bounds__(256) void fc_gemm_kernel(const float* __restrict__ enc,
                                                      const u16* __restrict__ Wb,
                                                      const float* __restrict__ bias,
                                                      u16* __restrict__ red) {
    constexpr int LDA = 40;                 // padded LDS stride (bf16 elems)
    __shared__ u16 As[2][128 * LDA];
    __shared__ u16 Bs[2][128 * LDA];

    const int tid  = threadIdx.x;
    const int m0   = blockIdx.y * 128;
    const int n0   = blockIdx.x * 128;
    const int w    = tid >> 5, lane = tid & 31;
    const int wm   = w >> 2, wn = w & 3;    // wave grid 2x4
    const int ln   = lane & 15, lh = lane >> 4;

    const int srow  = tid >> 1;             // staging row 0..127
    const int shalf = (tid & 1) * 16;       // k half 0/16

    const float* gA = enc + (size_t)(m0 + srow) * K2 + shalf;
    const u16*   gB = Wb  + (size_t)(n0 + srow) * K2 + shalf;

    v8f vz = {0.f, 0.f, 0.f, 0.f, 0.f, 0.f, 0.f, 0.f};
    v8f acc[4][2];
    for (int i = 0; i < 4; ++i)
        for (int j = 0; j < 2; ++j) acc[i][j] = vz;

    float4 a0, a1, a2, a3;                  // in-flight A tile (fp32)

    auto loadA = [&](int k0) {
        const float4* p = reinterpret_cast<const float4*>(gA + k0);
        a0 = p[0]; a1 = p[1]; a2 = p[2]; a3 = p[3];
    };
    auto writeA = [&](int buf) {
        uint4 p0 = make_uint4(pk2(a0.x, a0.y), pk2(a0.z, a0.w),
                              pk2(a1.x, a1.y), pk2(a1.z, a1.w));
        uint4 p1 = make_uint4(pk2(a2.x, a2.y), pk2(a2.z, a2.w),
                              pk2(a3.x, a3.y), pk2(a3.z, a3.w));
        uint4* sa = reinterpret_cast<uint4*>(&As[buf][srow * LDA + shalf]);
        sa[0] = p0; sa[1] = p1;
    };
    auto stageB = [&](int buf, int k0) {
#if defined(HAVE_ASYNC_LDS)
        // async DMA: cache -> LDS, no VGPR round trip; tracked by ASYNCcnt
        __builtin_amdgcn_global_load_async_to_lds_b128(
            (AS1 v4i*)(gB + k0), (AS3 v4i*)&Bs[buf][srow * LDA + shalf], 0, 0);
        __builtin_amdgcn_global_load_async_to_lds_b128(
            (AS1 v4i*)(gB + k0 + 8), (AS3 v4i*)&Bs[buf][srow * LDA + shalf + 8], 0,
            0);
#else
        const uint4* g = reinterpret_cast<const uint4*>(gB + k0);
        uint4 b0 = g[0], b1 = g[1];
        uint4* sb = reinterpret_cast<uint4*>(&Bs[buf][srow * LDA + shalf]);
        sb[0] = b0; sb[1] = b1;
#endif
    };

    // prologue: stage tile 0 into buffer 0
    loadA(0);
    stageB(0, 0);
    writeA(0);
    ASYNC_WAIT();
    __syncthreads();

    constexpr int NIT = K2 / 32;
    for (int it = 0; it < NIT; ++it) {
        const int cur = it & 1, nxt = cur ^ 1;
        const bool hasNext = (it + 1) < NIT;
        const int k1 = (it + 1) * 32;

        if (hasNext) {
            stageB(nxt, k1);   // async loads issue immediately
            loadA(k1);         // fp32 A tile -> regs (overlaps compute below)
        }

        v16bf af[4], bfr[2];
        for (int i = 0; i < 4; ++i)
            af[i] = frag_a16(&As[cur][(wm * 64 + i * 16 + ln) * LDA + lh * 16]);
        for (int j = 0; j < 2; ++j)
            bfr[j] = frag_b16(&Bs[cur][(wn * 32 + j * 16 + ln) * LDA + lh * 16]);
        for (int i = 0; i < 4; ++i)
            for (int j = 0; j < 2; ++j)
                acc[i][j] = wmma_bf16(af[i], bfr[j], acc[i][j]);

        if (hasNext) writeA(nxt);   // convert + ds_store into the other buffer
        ASYNC_WAIT();
        __syncthreads();
    }

    // epilogue: + bias, store bf16 reduced [M, H]
    for (int j = 0; j < 2; ++j) {
        int n = n0 + wn * 32 + j * 16 + ln;
        float bv = bias[n];
        for (int i = 0; i < 4; ++i) {
            int mbase = m0 + wm * 64 + i * 16 + lh * 8;
            for (int r = 0; r < 8; ++r)
                red[(size_t)(mbase + r) * Hd + n] = f2bf(acc[i][j][r] + bv);
        }
    }
}

// ---------------- kernel 3: scoresT[b,l,s] = sum_h red[b,s,h]*hid[l,b,h] ------
// Per (b, s-block of 32): 8 waves = 4 l-tiles x 2 s-tiles, K = H = 1024.
__global__ __launch_bounds__(256) void scores_gemm_kernel(const u16* __restrict__ hid,
                                                          const u16* __restrict__ red,
                                                          float* __restrict__ scoresT) {
    const int b  = blockIdx.y;
    const int s0 = blockIdx.x * 32;
    const int tid = threadIdx.x, w = tid >> 5, lane = tid & 31;
    const int wl = w >> 1, wsx = w & 1;
    const int ln = lane & 15, lh = lane >> 4;

    v8f acc = {0.f, 0.f, 0.f, 0.f, 0.f, 0.f, 0.f, 0.f};
    const u16* arow = hid + (size_t)(wl * 16 + ln) * (Bb * Hd) + (size_t)b * Hd + lh * 16;
    const int  sI   = s0 + wsx * 16 + ln;
    const u16* brow = red + ((size_t)b * Ss + sI) * Hd + lh * 16;

    for (int h0 = 0; h0 < Hd; h0 += 32) {
        v16bf a  = frag_a16(arow + h0);
        v16bf bb = frag_b16(brow + h0);
        acc = wmma_bf16(a, bb, acc);
    }
    for (int r = 0; r < 8; ++r) {
        int l = wl * 16 + lh * 8 + r;
        scoresT[((size_t)b * Ll + l) * Ss + sI] = acc[r];
    }
}

// ---------------- kernel 4: softmax over s; write weights + bf16 transpose ----
__global__ __launch_bounds__(256) void softmax_kernel(const float* __restrict__ scoresT,
                                                      float* __restrict__ wout,
                                                      u16* __restrict__ wT) {
    const int bl = blockIdx.x;              // b*L + l
    const int b = bl >> 6, l = bl & 63;
    const float* row = scoresT + (size_t)bl * Ss;
    __shared__ float sred[256];
    const int tid = threadIdx.x;

    float x[16];
    float mx = -3.402823466e+38f;
    for (int i = 0; i < 16; ++i) {
        x[i] = row[i * 256 + tid];
        mx = fmaxf(mx, x[i]);
    }
    sred[tid] = mx; __syncthreads();
    for (int off = 128; off > 0; off >>= 1) {
        if (tid < off) sred[tid] = fmaxf(sred[tid], sred[tid + off]);
        __syncthreads();
    }
    mx = sred[0]; __syncthreads();

    float sum = 0.f;
    for (int i = 0; i < 16; ++i) { x[i] = __expf(x[i] - mx); sum += x[i]; }
    sred[tid] = sum; __syncthreads();
    for (int off = 128; off > 0; off >>= 1) {
        if (tid < off) sred[tid] += sred[tid + off];
        __syncthreads();
    }
    float inv = 1.0f / sred[0];

    for (int i = 0; i < 16; ++i) {
        int s = i * 256 + tid;
        float v = x[i] * inv;
        wout[((size_t)b * Ss + s) * Ll + l] = v;   // [B,S,L] reference layout
        wT[(size_t)bl * Ss + s] = f2bf(v);         // [B,L,S] bf16 for next GEMM
    }
}

// ---------------- kernel 5: context[b,l,h] = sum_s wT[b,l,s]*red[b,s,h] -------
// M = L = 64, N-tile 64 of H, K = S, K-step 32; B staged transposed in LDS.
__global__ __launch_bounds__(256) void context_gemm_kernel(const u16* __restrict__ wT,
                                                           const u16* __restrict__ red,
                                                           float* __restrict__ ctx) {
    constexpr int LDB = 40;
    __shared__ u16 Bs[64 * LDB];            // [h_local][s_local]
    const int b  = blockIdx.y;
    const int h0 = blockIdx.x * 64;
    const int tid = threadIdx.x, w = tid >> 5, lane = tid & 31;
    const int wm = w >> 1, wn = w & 1;      // 4 l-tiles x 2 h-chunks
    const int ln = lane & 15, lh = lane >> 4;

    const int srow = tid >> 3;              // s_local 0..31
    const int hg   = (tid & 7) * 8;         // h_local group base

    v8f vz = {0.f, 0.f, 0.f, 0.f, 0.f, 0.f, 0.f, 0.f};
    v8f acc[2] = {vz, vz};
    const u16* abase = wT + ((size_t)b * Ll + wm * 16 + ln) * Ss + lh * 16;

    for (int s0 = 0; s0 < Ss; s0 += 32) {
        // stage reduced [32 s x 64 h] transposed into LDS
        uint4 q = *reinterpret_cast<const uint4*>(
            red + ((size_t)b * Ss + s0 + srow) * Hd + h0 + hg);
        union { uint4 v; u16 u[8]; } cv; cv.v = q;
        for (int i = 0; i < 8; ++i) Bs[(hg + i) * LDB + srow] = cv.u[i];
        __syncthreads();

        v16bf a = frag_a16(abase + s0);
        for (int j = 0; j < 2; ++j) {
            v16bf bb = frag_b16(&Bs[(wn * 32 + j * 16 + ln) * LDB + lh * 16]);
            acc[j] = wmma_bf16(a, bb, acc[j]);
        }
        __syncthreads();
    }
    for (int j = 0; j < 2; ++j) {
        int h = h0 + wn * 32 + j * 16 + ln;
        for (int r = 0; r < 8; ++r) {
            int l = wm * 16 + lh * 8 + r;
            ctx[((size_t)b * Ll + l) * Hd + h] = acc[j][r];
        }
    }
}

// ---------------- launch ----------------
extern "C" void kernel_launch(void* const* d_in, const int* in_sizes, int n_in,
                              void* d_out, int out_size, void* d_ws, size_t ws_size,
                              hipStream_t stream) {
    const float* hid  = (const float*)d_in[0];   // [L,B,H]
    const float* enc  = (const float*)d_in[1];   // [B,S,2H]
    const float* Wf   = (const float*)d_in[2];   // [H,2H]
    const float* bias = (const float*)d_in[3];   // [H]

    float* ctx_out = (float*)d_out;                          // [B,L,H]
    float* w_out   = (float*)d_out + (size_t)Bb * Ll * Hd;   // [B,S,L]

    // workspace layout (bytes)
    char* ws = (char*)d_ws;
    u16*   red_bf  = (u16*)(ws);                                        // 64 MB
    size_t off = (size_t)Bb * Ss * Hd * 2;
    u16*   Wbf     = (u16*)(ws + off);  off += (size_t)Hd * K2 * 2;     // +4 MB
    u16*   hidbf   = (u16*)(ws + off);  off += (size_t)Ll * Bb * Hd * 2;// +1 MB
    float* scoresT = (float*)(ws + off); off += (size_t)Bb * Ll * Ss * 4;// +8 MB
    u16*   wTbf    = (u16*)(ws + off);                                   // +4 MB

    convert_kernel<<<2048, 256, 0, stream>>>(Wf, Wbf, Hd * K2, hid, hidbf,
                                             Ll * Bb * Hd);
    fc_gemm_kernel<<<dim3(Hd / 128, (Bb * Ss) / 128), 256, 0, stream>>>(
        enc, Wbf, bias, red_bf);
    scores_gemm_kernel<<<dim3(Ss / 32, Bb), 256, 0, stream>>>(hidbf, red_bf, scoresT);
    softmax_kernel<<<Bb * Ll, 256, 0, stream>>>(scoresT, w_out, wTbf);
    context_gemm_kernel<<<dim3(Hd / 64, Bb), 256, 0, stream>>>(wTbf, red_bf, ctx_out);
}
